// NPairDiscriminator_36429912605242
// MI455X (gfx1250) — matrix operand order repaired and checked by reading, not compiled, not run on hardware
//
#include <hip/hip_runtime.h>
#include <cstdint>

typedef __attribute__((ext_vector_type(16))) _Float16 v16h;
typedef __attribute__((ext_vector_type(8)))  float    v8f;

#define NROWS 8192
#define DIM   128
#define TILE  128

union F16Frag { v16h v; uint4 u[2]; };

// ---------------------------------------------------------------------------
// Kernel 1: h = fc2(elu(fc1(z))); n = h / max(||h||, eps); store n as f16.
// One block of 128 threads handles 16 rows. Weights stay hot in L2 (128 KB).
// ---------------------------------------------------------------------------
__global__ __launch_bounds__(128) void project_norm_kernel(
    const float* __restrict__ z, const float* __restrict__ w1,
    const float* __restrict__ b1, const float* __restrict__ w2,
    const float* __restrict__ b2, _Float16* __restrict__ nout)
{
    __shared__ float zrow[DIM];
    __shared__ float tbuf[DIM];
    __shared__ float red[DIM];
    const int k = threadIdx.x;
    const int row0 = blockIdx.x * 16;

    for (int r = 0; r < 16; ++r) {
        const int row = row0 + r;
        zrow[k] = z[(size_t)row * DIM + k];
        __syncthreads();

        // fc1 row k
        float acc = b1[k];
        const float* wr = w1 + (size_t)k * DIM;
        #pragma unroll 8
        for (int d = 0; d < DIM; ++d) acc = fmaf(zrow[d], wr[d], acc);
        // ELU (alpha = 1)
        tbuf[k] = acc > 0.0f ? acc : (expf(acc) - 1.0f);
        __syncthreads();

        // fc2 row k
        float acc2 = b2[k];
        const float* wr2 = w2 + (size_t)k * DIM;
        #pragma unroll 8
        for (int d = 0; d < DIM; ++d) acc2 = fmaf(tbuf[d], wr2[d], acc2);

        red[k] = acc2 * acc2;
        __syncthreads();
        for (int s = 64; s > 0; s >>= 1) {
            if (k < s) red[k] += red[k + s];
            __syncthreads();
        }
        const float rn = 1.0f / fmaxf(sqrtf(red[0]), 1e-12f);
        nout[(size_t)row * DIM + k] = (_Float16)(acc2 * rn);
        __syncthreads();
    }
}

// ---------------------------------------------------------------------------
// Kernel 2: zero the atomically-accumulated reduction vectors.
// ---------------------------------------------------------------------------
__global__ void zero_stats_kernel(float* __restrict__ p, int n)
{
    int i = blockIdx.x * 256 + threadIdx.x;
    if (i < n) p[i] = 0.0f;
}

// ---------------------------------------------------------------------------
// Kernel 3: fused similarity GEMM + exp(S/tau) + row/col sums + diagonals.
// blockIdx.y = matrix id: 0 -> n1@n1^T (sym), 1 -> n2@n2^T (sym), 2 -> n1@n2^T.
// 128x128 tile, full K=128. 256 threads = 8 waves arranged 2x4; each wave
// computes a 64x32 region = 4x2 WMMA accumulators over 4 k-steps of 32.
// A/B tiles staged via async global->LDS (ASYNCcnt); ALL fragments are
// hoisted into VGPRs (192 regs) before issuing 32 back-to-back WMMAs.
// Epilogue reduces with in-wave shuffles -> conflict-free LDS atomics.
// ---------------------------------------------------------------------------
__global__ __launch_bounds__(256) void sim_kernel(
    const _Float16* __restrict__ n1, const _Float16* __restrict__ n2,
    float* __restrict__ R11, float* __restrict__ R22,
    float* __restrict__ Brow, float* __restrict__ Bcol,
    float* __restrict__ d11, float* __restrict__ d22, float* __restrict__ d12)
{
    const int mat = blockIdx.y;
    const int tm  = blockIdx.x / (NROWS / TILE);
    const int tn  = blockIdx.x % (NROWS / TILE);
    if (mat < 2 && tn < tm) return;   // symmetric: upper triangle only

    const _Float16* A = (mat == 1) ? n2 : n1;
    const _Float16* B = (mat == 0) ? n1 : n2;
    float* rowT  = (mat == 0) ? R11 : (mat == 1) ? R22 : Brow;
    float* colT  = (mat == 0) ? R11 : (mat == 1) ? R22 : Bcol;
    float* diagT = (mat == 0) ? d11 : (mat == 1) ? d22 : d12;
    const bool doCol = (mat == 2) || (tn > tm);

    // 32 KB A tile + 32 KB B tile (rows contiguous: each tile is a 32 KB blob)
    __shared__ uint4 lA4[TILE * DIM / 8];
    __shared__ uint4 lB4[TILE * DIM / 8];

    const int t = threadIdx.x;

    // ---- async copy: 2 x 32 KB contiguous, 128 B per thread per tile ----
    {
        const _Float16* gA = A + (size_t)tm * TILE * DIM;
        const _Float16* gB = B + (size_t)tn * TILE * DIM;
        const unsigned lA0 = (unsigned)(uintptr_t)lA4;
        const unsigned lB0 = (unsigned)(uintptr_t)lB4;
        #pragma unroll
        for (int it = 0; it < 8; ++it) {
            unsigned off = (unsigned)(it * 4096 + t * 16);
            unsigned la = lA0 + off;
            unsigned lb = lB0 + off;
            asm volatile("global_load_async_to_lds_b128 %0, %1, %2"
                         :: "v"(la), "v"(off), "s"(gA) : "memory");
            asm volatile("global_load_async_to_lds_b128 %0, %1, %2"
                         :: "v"(lb), "v"(off), "s"(gB) : "memory");
        }
        asm volatile("s_wait_asynccnt 0" ::: "memory");
    }
    __syncthreads();

    const int lane = t & 31;
    const int wave = t >> 5;
    const int wr = wave >> 2;        // 0..1  -> 64-row band
    const int wc = wave & 3;         // 0..3  -> 32-col band
    const int l16 = lane & 15;
    const int hi  = lane >> 4;

    // ---- hoist ALL fragments into VGPRs, then issue WMMAs back-to-back ----
    F16Frag af[4][4];   // [ks][mi]  A: row (wr*64+mi*16+l16), K 16 halfs +hi*16
    F16Frag bf[4][2];   // [ks][nj]  B: col (wc*32+nj*16+l16), K 16 halfs +hi*16
    #pragma unroll
    for (int ks = 0; ks < 4; ++ks) {
        #pragma unroll
        for (int nj = 0; nj < 2; ++nj) {
            int base = (wc * 32 + nj * 16 + l16) * 16 + ks * 4 + hi * 2; // uint4 units
            bf[ks][nj].u[0] = lB4[base];
            bf[ks][nj].u[1] = lB4[base + 1];
        }
        #pragma unroll
        for (int mi = 0; mi < 4; ++mi) {
            int base = (wr * 64 + mi * 16 + l16) * 16 + ks * 4 + hi * 2; // uint4 units
            af[ks][mi].u[0] = lA4[base];
            af[ks][mi].u[1] = lA4[base + 1];
        }
    }

    v8f acc[4][2];
    #pragma unroll
    for (int mi = 0; mi < 4; ++mi)
        #pragma unroll
        for (int nj = 0; nj < 2; ++nj) acc[mi][nj] = (v8f)(0.0f);

    #pragma unroll
    for (int ks = 0; ks < 4; ++ks)
        #pragma unroll
        for (int mi = 0; mi < 4; ++mi)
            #pragma unroll
            for (int nj = 0; nj < 2; ++nj)
                acc[mi][nj] = __builtin_amdgcn_wmma_f32_16x16x32_f16(
                    false, af[ks][mi].v, false, bf[ks][nj].v,
                    (short)0, acc[mi][nj], false, false);

    __syncthreads();                       // tiles consumed; reuse lA4 for sums
    float* srow = (float*)lA4;
    float* scol = srow + TILE;
    if (t < TILE) { srow[t] = 0.0f; scol[t] = 0.0f; }
    __syncthreads();

    // exp(S/tau) = exp2(S * log2(e)/tau), tau = 0.5
    const float C = 2.8853900817779268f;
    float colpart[2] = {0.0f, 0.0f};       // col = wc*32 + nj*16 + l16

    #pragma unroll
    for (int mi = 0; mi < 4; ++mi) {
        float rowpart[8];
        #pragma unroll
        for (int v = 0; v < 8; ++v) rowpart[v] = 0.0f;

        #pragma unroll
        for (int nj = 0; nj < 2; ++nj) {
            const int n_local = wc * 32 + nj * 16 + l16;
            #pragma unroll
            for (int v = 0; v < 8; ++v) {
                const int m_local = wr * 64 + mi * 16 + hi * 8 + v;
                float e = exp2f(acc[mi][nj][v] * C);
                rowpart[v] += e;
                colpart[nj] += e;
                if (tm == tn && m_local == n_local)
                    diagT[tm * TILE + m_local] = e;
            }
        }
        // butterfly-reduce row partials across the 16 lanes of each half-wave
        #pragma unroll
        for (int m = 1; m < 16; m <<= 1) {
            #pragma unroll
            for (int v = 0; v < 8; ++v)
                rowpart[v] += __shfl_xor(rowpart[v], m, 32);
        }
        if (l16 == 0) {                    // 2 lanes, 8 distinct rows each
            #pragma unroll
            for (int v = 0; v < 8; ++v)
                atomicAdd(&srow[wr * 64 + mi * 16 + hi * 8 + v], rowpart[v]);
        }
    }
    if (doCol) {
        #pragma unroll
        for (int nj = 0; nj < 2; ++nj) {
            colpart[nj] += __shfl_xor(colpart[nj], 16, 32);  // fold hi half
            if (hi == 0)                   // 16 lanes, distinct columns
                atomicAdd(&scol[wc * 32 + nj * 16 + l16], colpart[nj]);
        }
    }
    __syncthreads();

    if (t < TILE) {
        atomicAdd(&rowT[tm * TILE + t], srow[t]);
        if (doCol) atomicAdd(&colT[tn * TILE + t], scol[t]);
    }
}

// ---------------------------------------------------------------------------
// Kernel 4: loss = mean_i 0.5*(l1[i] + l2[i])
//   l1 = -log(d12 / (R11 + Brow - d11)),  l2 = -log(d12 / (R22 + Bcol - d22))
// ---------------------------------------------------------------------------
__global__ __launch_bounds__(256) void finalize_kernel(
    const float* __restrict__ R11, const float* __restrict__ R22,
    const float* __restrict__ Brow, const float* __restrict__ Bcol,
    const float* __restrict__ d11, const float* __restrict__ d22,
    const float* __restrict__ d12, float* __restrict__ out)
{
    __shared__ float red[256];
    float acc = 0.0f;
    for (int i = threadIdx.x; i < NROWS; i += 256) {
        float den1 = R11[i] + Brow[i] - d11[i];
        float den2 = R22[i] + Bcol[i] - d22[i];
        float l1 = -logf(d12[i] / den1);
        float l2 = -logf(d12[i] / den2);
        acc += 0.5f * (l1 + l2);
    }
    red[threadIdx.x] = acc;
    __syncthreads();
    for (int s = 128; s > 0; s >>= 1) {
        if (threadIdx.x < s) red[threadIdx.x] += red[threadIdx.x + s];
        __syncthreads();
    }
    if (threadIdx.x == 0) out[0] = red[0] / (float)NROWS;
}

// ---------------------------------------------------------------------------
extern "C" void kernel_launch(void* const* d_in, const int* in_sizes, int n_in,
                              void* d_out, int out_size, void* d_ws, size_t ws_size,
                              hipStream_t stream)
{
    const float* z1 = (const float*)d_in[0];
    const float* z2 = (const float*)d_in[1];
    const float* w1 = (const float*)d_in[2];
    const float* b1 = (const float*)d_in[3];
    const float* w2 = (const float*)d_in[4];
    const float* b2 = (const float*)d_in[5];
    float* out = (float*)d_out;

    char* ws = (char*)d_ws;
    _Float16* n1h = (_Float16*)ws;                       // 2 MB
    _Float16* n2h = n1h + (size_t)NROWS * DIM;           // 2 MB
    float* stats  = (float*)(ws + 4u * 1024u * 1024u);
    float* R11  = stats + 0 * NROWS;
    float* R22  = stats + 1 * NROWS;
    float* Brow = stats + 2 * NROWS;
    float* Bcol = stats + 3 * NROWS;
    float* d11  = stats + 4 * NROWS;
    float* d22  = stats + 5 * NROWS;
    float* d12  = stats + 6 * NROWS;

    // zero the 4 accumulated vectors (d* are fully overwritten)
    zero_stats_kernel<<<(4 * NROWS + 255) / 256, 256, 0, stream>>>(stats, 4 * NROWS);

    project_norm_kernel<<<NROWS / 16, 128, 0, stream>>>(z1, w1, b1, w2, b2, n1h);
    project_norm_kernel<<<NROWS / 16, 128, 0, stream>>>(z2, w1, b1, w2, b2, n2h);

    dim3 grid((NROWS / TILE) * (NROWS / TILE), 3);
    sim_kernel<<<grid, 256, 0, stream>>>(n1h, n2h, R11, R22, Brow, Bcol, d11, d22, d12);

    finalize_kernel<<<1, 256, 0, stream>>>(R11, R22, Brow, Bcol, d11, d22, d12, out);
}
